// MultiHeadAttention_22634477650074
// MI455X (gfx1250) — matrix operand order repaired and checked by reading, not compiled, not run on hardware
//
#include <hip/hip_runtime.h>
#include <hip/hip_bf16.h>

// MI455X / gfx1250 wave32 WMMA fused MHA forward.
// f16 operands, f32 accumulation (v_wmma_f32_16x16x32_f16).
// Attention stages K/V tiles into LDS with the Tensor Data Mover
// (tensor_load_to_lds + s_wait_tensorcnt), double-buffered, shared by
// 4 query-tile waves per workgroup.

#define B_  2
#define S_  2048
#define D_  1024
#define H_  16
#define HD_ 64

typedef __attribute__((ext_vector_type(16))) _Float16 v16h;
typedef __attribute__((ext_vector_type(8)))  _Float16 v8h;
typedef __attribute__((ext_vector_type(8)))  float    v8f;
typedef __attribute__((ext_vector_type(4)))  unsigned int v4u;
typedef __attribute__((ext_vector_type(8)))  int      v8i;
typedef __attribute__((ext_vector_type(4)))  int      v4i;

__device__ __forceinline__ v8f wmma_f16(const v16h& a, const v16h& b, const v8f& c) {
  return __builtin_amdgcn_wmma_f32_16x16x32_f16(false, a, false, b, (short)0, c,
                                                false, false);
}

__device__ __forceinline__ v16h frag_a(const _Float16* lo_p, const _Float16* hi_p) {
  v8h lo = *(const v8h*)lo_p;
  v8h hi = *(const v8h*)hi_p;
  v16h a;
#pragma unroll
  for (int i = 0; i < 8; ++i) { a[i] = lo[i]; a[8 + i] = hi[i]; }
  return a;
}

// ---------------------------------------------------------------------------
// TDM: DMA a 2-D f16 tile (tile_x fast-dim elems  x  tile_y rows, row stride
// stride_x elems) from global memory into LDS at byte offset lds_off.
// Descriptor layout per CDNA5 ISA (D# groups 0/1).
// ---------------------------------------------------------------------------
__device__ __forceinline__ void tdm_load_2d(unsigned lds_off, const void* gaddr,
                                            unsigned tile_x, unsigned tile_y,
                                            unsigned stride_x) {
  const unsigned long long ga = (unsigned long long)(size_t)gaddr;
  v4u g0;
  g0[0] = 1u;                                        // count=1 (valid), user mode
  g0[1] = lds_off;                                   // lds_addr (bytes)
  g0[2] = (unsigned)(ga & 0xffffffffu);              // global_addr[31:0]
  g0[3] = (unsigned)((ga >> 32) & 0x01ffffffu)       // global_addr[56:32]
        | (2u << 30);                                // type=2 ("image")
  v8i g1;
  g1[0] = (int)(1u << 16);                           // data_size=1 -> 2 bytes
  g1[1] = (int)((tile_x & 0xffffu) << 16);           // tensor_dim0[15:0]
  g1[2] = (int)((tile_y & 0xffffu) << 16);           // tensor_dim1[15:0]
  g1[3] = (int)((tile_x & 0xffffu) << 16);           // tile_dim0
  g1[4] = (int)(tile_y & 0xffffu);                   // tile_dim1 (tile_dim2=0)
  g1[5] = (int)stride_x;                             // tensor_dim0_stride[31:0]
  g1[6] = 0;
  g1[7] = 0;
  const v4i z4 = {0, 0, 0, 0};
#if __clang_major__ >= 23
  const v8i z8 = {0, 0, 0, 0, 0, 0, 0, 0};
  __builtin_amdgcn_tensor_load_to_lds(g0, g1, z4, z4, z8, 0);
#else
  __builtin_amdgcn_tensor_load_to_lds(g0, g1, z4, z4, 0);
#endif
}

__device__ __forceinline__ unsigned lds_offset(const void* p) {
  return (unsigned)(size_t)p;   // flat LDS pointer: low 32 bits = LDS byte addr
}

// ---------------------------------------------------------------------------
// Pre-pass A: x (f32) -> xh (f16), 8 elems/thread.
// ---------------------------------------------------------------------------
__global__ __launch_bounds__(256)
void xconv_kernel(const float* __restrict__ x, _Float16* __restrict__ xh) {
  const size_t i = ((size_t)blockIdx.x * 256 + threadIdx.x) * 8;
  const float4 a = *(const float4*)(x + i);
  const float4 b = *(const float4*)(x + i + 4);
  v8h o;
  o[0] = (_Float16)a.x; o[1] = (_Float16)a.y; o[2] = (_Float16)a.z; o[3] = (_Float16)a.w;
  o[4] = (_Float16)b.x; o[5] = (_Float16)b.y; o[6] = (_Float16)b.z; o[7] = (_Float16)b.w;
  *(v8h*)(xh + i) = o;
}

// ---------------------------------------------------------------------------
// Pre-pass B: weight f32 [k][n] -> f16 transposed [n][k] (LDS tile transpose).
// grid = (D/32, D/32, 4); block = (32, 8).
// ---------------------------------------------------------------------------
__global__ __launch_bounds__(256)
void wtrans_kernel(const float* __restrict__ wq, const float* __restrict__ wk,
                   const float* __restrict__ wv, const float* __restrict__ wo,
                   _Float16* __restrict__ wqT, _Float16* __restrict__ wkT,
                   _Float16* __restrict__ wvT, _Float16* __restrict__ woT) {
  __shared__ float tile[32][33];
  const int z = blockIdx.z;
  const float* w = (z == 0) ? wq : (z == 1) ? wk : (z == 2) ? wv : wo;
  _Float16*   wT = (z == 0) ? wqT : (z == 1) ? wkT : (z == 2) ? wvT : woT;
  const int k0 = blockIdx.x * 32, n0 = blockIdx.y * 32;
  const int tx = threadIdx.x, ty = threadIdx.y;
#pragma unroll
  for (int j = 0; j < 4; ++j)
    tile[ty + 8 * j][tx] = w[(size_t)(k0 + ty + 8 * j) * D_ + n0 + tx];
  __syncthreads();
#pragma unroll
  for (int j = 0; j < 4; ++j)
    wT[(size_t)(n0 + ty + 8 * j) * D_ + k0 + tx] = (_Float16)tile[tx][ty + 8 * j];
}

// ---------------------------------------------------------------------------
// Kernel 1: QKV projection + bias + RoPE, one wave per 16x64 tile (one head).
// grid = (B*S/16, D/64, 3); block = 32.  Q,K -> [B][H][S][HD]; V -> [B][H][HD][S].
// ---------------------------------------------------------------------------
__global__ __launch_bounds__(32)
void qkv_rope_kernel(const _Float16* __restrict__ xh,
                     const float* __restrict__ freqs,     // [S][HD/2][2]
                     const _Float16* __restrict__ wqT, const _Float16* __restrict__ wkT,
                     const _Float16* __restrict__ wvT,
                     const float* __restrict__ bq, const float* __restrict__ bk,
                     const float* __restrict__ bv,
                     _Float16* __restrict__ qh, _Float16* __restrict__ kh,
                     _Float16* __restrict__ vt) {
  const int lane = threadIdx.x;
  const int lr   = lane & 15;
  const int half = lane >> 4;
  const int m0   = blockIdx.x * 16;
  const int h    = blockIdx.y;
  const int n0   = h * HD_;
  const int mat  = blockIdx.z;          // 0=Q 1=K 2=V

  const _Float16* wT   = (mat == 0) ? wqT : (mat == 1) ? wkT : wvT;
  const float*    bias = (mat == 0) ? bq  : (mat == 1) ? bk  : bv;

  const _Float16* arow = xh + (size_t)(m0 + lr) * D_;
  v8f acc[4] = {};
  for (int k0 = 0; k0 < D_; k0 += 32) {
    const v16h a = frag_a(arow + k0 + 16 * half, arow + k0 + 16 * half + 8);
#pragma unroll
    for (int c = 0; c < 4; ++c) {
      const v16h b = *(const v16h*)(wT + (size_t)(n0 + c * 16 + lr) * D_ +
                                    k0 + 16 * half);
      acc[c] = wmma_f16(a, b, acc[c]);
    }
  }

  const float qscale = (mat == 0) ? 0.125f : 1.0f;   // 1/sqrt(HD)
#pragma unroll
  for (int c = 0; c < 4; ++c) {
    const int hd = c * 16 + lr;
    const float bias_v = bias[n0 + hd];
    float vb[8];
#pragma unroll
    for (int r = 0; r < 8; ++r) vb[r] = acc[c][r] + bias_v;

    if (mat < 2) {
      _Float16* dst = (mat == 0) ? qh : kh;
#pragma unroll
      for (int r = 0; r < 8; ++r) {
        const int t = m0 + r + 8 * half;
        const int bb = t / S_, ss = t % S_;
        const float p  = __shfl_xor(vb[r], 1, 32);   // rotary partner = lane^1
        const size_t f = (size_t)ss * HD_ + (hd & ~1);
        const float cs = freqs[f], sn = freqs[f + 1];
        float o = (hd & 1) ? (p * sn + vb[r] * cs)
                           : (vb[r] * cs - p * sn);
        o *= qscale;
        dst[(((size_t)bb * H_ + h) * S_ + ss) * HD_ + hd] = (_Float16)o;
      }
    } else {
#pragma unroll
      for (int r = 0; r < 8; ++r) {
        const int t = m0 + r + 8 * half;
        const int bb = t / S_, ss = t % S_;
        vt[(((size_t)bb * H_ + h) * HD_ + hd) * S_ + ss] = (_Float16)vb[r];
      }
    }
  }
}

// ---------------------------------------------------------------------------
// Kernel 2: flash attention. Block = 4 waves = 4 query tiles of one (b,h).
// K (32x64) and V^T (64x32) tiles are DMA'd into double-buffered LDS by the
// TDM (wave 0 issues, all waves consume). grid = (S/64, B*H); block = 128.
// ---------------------------------------------------------------------------
__global__ __launch_bounds__(128)
void attn_kernel(const _Float16* __restrict__ qh,
                 const _Float16* __restrict__ kh,
                 const _Float16* __restrict__ vt,   // [B][H][HD][S]
                 _Float16* __restrict__ oh) {
  __shared__ __align__(64) _Float16 lds_k[2][32 * HD_];   // [key][d]
  __shared__ __align__(64) _Float16 lds_v[2][HD_ * 32];   // [hd][key]
  __shared__ __align__(64) _Float16 pbuf[4][16 * 32];

  const int tid  = threadIdx.x;
  const int wid  = tid >> 5;
  const int lane = tid & 31;
  const int lr   = lane & 15;
  const int half = lane >> 4;
  const int q0   = (blockIdx.x * 4 + wid) * 16;
  const int bhid = blockIdx.y;
  const int bb   = bhid / H_, hh = bhid % H_;

  const _Float16* qp  = qh + (size_t)bhid * S_ * HD_;
  const _Float16* kp  = kh + (size_t)bhid * S_ * HD_;
  const _Float16* vtp = vt + (size_t)bhid * HD_ * S_;

  // Q A-fragments, d-chunks [0,32) and [32,64) (pre-scaled by 1/8).
  const _Float16* qrow = qp + (size_t)(q0 + lr) * HD_;
  const v16h aq0 = frag_a(qrow + 16 * half,      qrow + 16 * half + 8);
  const v16h aq1 = frag_a(qrow + 32 + 16 * half, qrow + 40 + 16 * half);

  float m_st[8], l_st[8];
#pragma unroll
  for (int r = 0; r < 8; ++r) { m_st[r] = -3.0e38f; l_st[r] = 0.0f; }
  v8f o[4] = {};

  // Prologue: TDM-fill buffer 0 with the first K/V block.
  if (wid == 0) {
    tdm_load_2d(lds_offset(&lds_k[0][0]), kp,  HD_, 32, HD_);
    tdm_load_2d(lds_offset(&lds_v[0][0]), vtp, 32, HD_, S_);
  }

  int cur = 0;
  for (int kb = 0; kb < S_; kb += 32) {
    if (wid == 0) {
      if (kb + 32 < S_) {
        // Kick the next block into the other buffer, then wait until only
        // those 2 new TDM ops remain outstanding (TDM is in-order per wave).
        tdm_load_2d(lds_offset(&lds_k[cur ^ 1][0]), kp + (size_t)(kb + 32) * HD_,
                    HD_, 32, HD_);
        tdm_load_2d(lds_offset(&lds_v[cur ^ 1][0]), vtp + kb + 32,
                    32, HD_, S_);
        __builtin_amdgcn_s_wait_tensorcnt(2);
      } else {
        __builtin_amdgcn_s_wait_tensorcnt(0);
      }
    }
    __syncthreads();   // current buffer visible to all 4 waves

    const _Float16* K = lds_k[cur];
    const _Float16* V = lds_v[cur];

    // ---- scores: Q @ K^T for keys [kb, kb+32) as two 16x16 f32 tiles ----
    const v16h bk00 = *(const v16h*)(K + (size_t)lr * HD_ + 16 * half);
    const v16h bk01 = *(const v16h*)(K + (size_t)lr * HD_ + 32 + 16 * half);
    const v16h bk10 = *(const v16h*)(K + (size_t)(16 + lr) * HD_ + 16 * half);
    const v16h bk11 = *(const v16h*)(K + (size_t)(16 + lr) * HD_ + 32 + 16 * half);
    v8f s0 = {}, s1 = {};
    s0 = wmma_f16(aq0, bk00, s0);
    s0 = wmma_f16(aq1, bk01, s0);
    s1 = wmma_f16(aq0, bk10, s1);
    s1 = wmma_f16(aq1, bk11, s1);

    // ---- online softmax: row = r + 8*half, 16 columns striped over lanes ---
#pragma unroll
    for (int r = 0; r < 8; ++r) {
      float mx = fmaxf(s0[r], s1[r]);
      mx = fmaxf(mx, __shfl_xor(mx, 1, 32));
      mx = fmaxf(mx, __shfl_xor(mx, 2, 32));
      mx = fmaxf(mx, __shfl_xor(mx, 4, 32));
      mx = fmaxf(mx, __shfl_xor(mx, 8, 32));
      const float mnew  = fmaxf(m_st[r], mx);
      const float alpha = __expf(m_st[r] - mnew);
      const float p0 = __expf(s0[r] - mnew);
      const float p1 = __expf(s1[r] - mnew);
      float rs = p0 + p1;
      rs += __shfl_xor(rs, 1, 32);
      rs += __shfl_xor(rs, 2, 32);
      rs += __shfl_xor(rs, 4, 32);
      rs += __shfl_xor(rs, 8, 32);
      l_st[r] = l_st[r] * alpha + rs;
      m_st[r] = mnew;
      o[0][r] *= alpha; o[1][r] *= alpha; o[2][r] *= alpha; o[3][r] *= alpha;
      pbuf[wid][(r + 8 * half) * 32 + lr]      = (_Float16)p0;
      pbuf[wid][(r + 8 * half) * 32 + 16 + lr] = (_Float16)p1;
    }
    __syncthreads();

    // ---- P (16x32, C layout) reloaded as A-fragment from LDS ----
    const v16h ap = frag_a(pbuf[wid] + lr * 32 + 16 * half,
                           pbuf[wid] + lr * 32 + 8 + 16 * half);

    // ---- O += P @ V : V^T rows contiguous in key dimension ----
#pragma unroll
    for (int c = 0; c < 4; ++c) {
      const v16h bv = *(const v16h*)(V + (size_t)(c * 16 + lr) * 32 + 16 * half);
      o[c] = wmma_f16(ap, bv, o[c]);
    }
    __syncthreads();   // all reads of buf[cur] done before TDM overwrites it
    cur ^= 1;
  }

  // ---- normalize, store token-major f16 [B][S][D] ----
#pragma unroll
  for (int r = 0; r < 8; ++r) {
    const float inv = 1.0f / l_st[r];
    const int ss = q0 + r + 8 * half;
    _Float16* dst = oh + ((size_t)bb * S_ + ss) * D_ + hh * HD_;
#pragma unroll
    for (int c = 0; c < 4; ++c)
      dst[c * 16 + lr] = (_Float16)(o[c][r] * inv);
  }
}

// ---------------------------------------------------------------------------
// Kernel 3: output projection  out = oh @ wo + bo (fp32), 16x64 per wave.
// grid = (B*S/16, D/64); block = 32.
// ---------------------------------------------------------------------------
__global__ __launch_bounds__(32)
void oproj_kernel(const _Float16* __restrict__ oh,
                  const _Float16* __restrict__ woT,
                  const float* __restrict__ bo,
                  float* __restrict__ out) {
  const int lane = threadIdx.x;
  const int lr   = lane & 15;
  const int half = lane >> 4;
  const int m0   = blockIdx.x * 16;
  const int n0   = blockIdx.y * 64;

  const _Float16* arow = oh + (size_t)(m0 + lr) * D_;
  v8f acc[4] = {};
  for (int k0 = 0; k0 < D_; k0 += 32) {
    const v16h a = frag_a(arow + k0 + 16 * half, arow + k0 + 16 * half + 8);
#pragma unroll
    for (int c = 0; c < 4; ++c) {
      const v16h b = *(const v16h*)(woT + (size_t)(n0 + c * 16 + lr) * D_ +
                                    k0 + 16 * half);
      acc[c] = wmma_f16(a, b, acc[c]);
    }
  }

#pragma unroll
  for (int c = 0; c < 4; ++c) {
    const int n = n0 + c * 16 + lr;
    const float bias = bo[n];
#pragma unroll
    for (int r = 0; r < 8; ++r) {
      const int t = m0 + r + 8 * half;
      out[(size_t)t * D_ + n] = acc[c][r] + bias;
    }
  }
}

// ---------------------------------------------------------------------------
extern "C" void kernel_launch(void* const* d_in, const int* in_sizes, int n_in,
                              void* d_out, int out_size, void* d_ws, size_t ws_size,
                              hipStream_t stream) {
  const float* x     = (const float*)d_in[0];
  const float* freqs = (const float*)d_in[1];
  const float* wq    = (const float*)d_in[2];
  const float* bq    = (const float*)d_in[3];
  const float* wk    = (const float*)d_in[4];
  const float* bk    = (const float*)d_in[5];
  const float* wv    = (const float*)d_in[6];
  const float* bv    = (const float*)d_in[7];
  const float* wo    = (const float*)d_in[8];
  const float* bo    = (const float*)d_in[9];
  float*       out   = (float*)d_out;
  (void)in_sizes; (void)n_in; (void)out_size; (void)ws_size;

  const size_t tok = (size_t)B_ * S_ * D_;          // 4M (xh / oh)
  const size_t wsz = (size_t)D_ * D_;               // 1M per weight
  const size_t hsz = (size_t)B_ * H_ * S_ * HD_;    // 4M (qh/kh/vt)

  _Float16* xh  = (_Float16*)d_ws;
  _Float16* wqT = xh + tok;
  _Float16* wkT = wqT + wsz;
  _Float16* wvT = wkT + wsz;
  _Float16* woT = wvT + wsz;
  _Float16* qhb = woT + wsz;
  _Float16* khb = qhb + hsz;
  _Float16* vtb = khb + hsz;
  _Float16* ohb = vtb + hsz;

  xconv_kernel<<<dim3(tok / (256 * 8)), 256, 0, stream>>>(x, xh);
  wtrans_kernel<<<dim3(D_ / 32, D_ / 32, 4), dim3(32, 8), 0, stream>>>(
      wq, wk, wv, wo, wqT, wkT, wvT, woT);

  qkv_rope_kernel<<<dim3((B_ * S_) / 16, D_ / HD_, 3), 32, 0, stream>>>(
      xh, freqs, wqT, wkT, wvT, bq, bk, bv, qhb, khb, vtb);

  attn_kernel<<<dim3(S_ / 64, B_ * H_), 128, 0, stream>>>(qhb, khb, vtb, ohb);

  oproj_kernel<<<dim3((B_ * S_) / 16, D_ / 64), 32, 0, stream>>>(ohb, woT, bo, out);
}
